// NLBlock_40664750359453
// MI455X (gfx1250) — matrix operand order repaired and checked by reading, not compiled, not run on hardware
//
#include <hip/hip_runtime.h>
#include <hip/hip_bf16.h>
#include <stddef.h>

typedef __attribute__((ext_vector_type(16))) __bf16 v16bf;
typedef __attribute__((ext_vector_type(8)))  float  v8f;

namespace {

constexpr int kB = 16, kC = 256, kHW = 3136;
constexpr int kCg = 64, kCi = 32, kNp = 784, kNpPad = 800;
constexpr int kNtot = kB * kHW; // 50176
constexpr float kEps = 1e-5f;

// workspace layout (float offsets); total ~24.7 MB
constexpr size_t OFF_SCALE1  = 0;                                          // 256
constexpr size_t OFF_BIAS1   = 256;                                        // 256
constexpr size_t OFF_THETA   = 512;                                        // B*Ci*HW
constexpr size_t OFF_GPOOL   = OFF_THETA + (size_t)kB * kCi * kHW;         // B*Cg*NpPad
constexpr size_t OFF_PHIPOOL = OFF_GPOOL + (size_t)kB * kCg * kNpPad;      // B*Ci*NpPad
constexpr size_t OFF_SCALE2  = OFF_PHIPOOL + (size_t)kB * kCi * kNpPad;    // 64
constexpr size_t OFF_BIAS2   = OFF_SCALE2 + 64;                            // 64
constexpr size_t OFF_M       = OFF_BIAS2 + 64;                             // B*Ci*Cg
constexpr size_t OFF_Y       = OFF_M + (size_t)kB * kCi * kCg;             // B*Cg*HW
constexpr size_t OFF_PART    = OFF_Y + (size_t)kB * kCg * kHW;             // 784*128
constexpr size_t ZERO_BASE   = OFF_GPOOL;
constexpr size_t ZERO_LEN    = (size_t)kB * kCg * kNpPad + (size_t)kB * kCi * kNpPad;

__device__ inline v8f vzero8() {
  v8f z;
#pragma unroll
  for (int i = 0; i < 8; ++i) z[i] = 0.f;
  return z;
}

// e -> K mapping for the A-matrix (16x32 bf16) fragments as staged in this file (contiguous order, not the ISA table order):
// lane half h (0/1): k = h*16 + e
__device__ inline int a_k_of_e(int e, int half) {
  return (half << 4) + e;
}
// e -> K mapping for the B-matrix (32x16 bf16): k = half*16 + e

// Fragment-major LDS tiles: tile stored as [frag][lane 0..31][e 0..15],
// so each lane's fragment is 32 contiguous bytes -> 2x ds_load_b128.
__device__ inline v16bf ld_frag(const __bf16* tile, int frag, int lane) {
  return *(const v16bf*)(tile + (((frag << 5) + lane) << 4));
}

__device__ inline v8f wmma_bf16(v16bf a, v16bf b, v8f c) {
  return __builtin_amdgcn_wmma_f32_16x16x32_bf16(false, a, false, b, (short)0, c,
                                                 false, false);
}

} // namespace

// ---------------- K0: zero pooled buffers (pad lanes must be 0 for K3 GEMM) ----
__global__ void k0_zero(float* __restrict__ ws) {
  size_t i = (size_t)blockIdx.x * blockDim.x + threadIdx.x;
  if (i < ZERO_LEN) ws[ZERO_BASE + i] = 0.f;
}

// ---------------- K1: BN1 batch stats -> per-channel scale/bias ---------------
__global__ void __launch_bounds__(256) k1_bn1_stats(const float* __restrict__ x,
                                                    const float* __restrict__ gamma,
                                                    const float* __restrict__ beta,
                                                    float* __restrict__ ws) {
  int c = blockIdx.x, tid = threadIdx.x;
  float s = 0.f, q = 0.f;
  for (int b = 0; b < kB; ++b) {
    const float* p = x + ((size_t)(b * kC + c)) * kHW;
    for (int i = tid; i < kHW; i += 256) { float v = p[i]; s += v; q += v * v; }
  }
  __shared__ float rs[256], rq[256];
  rs[tid] = s; rq[tid] = q;
  __syncthreads();
  for (int off = 128; off > 0; off >>= 1) {
    if (tid < off) { rs[tid] += rs[tid + off]; rq[tid] += rq[tid + off]; }
    __syncthreads();
  }
  if (tid == 0) {
    float mean = rs[0] / (float)kNtot;
    float var = rq[0] / (float)kNtot - mean * mean;
    float sc = gamma[c] * rsqrtf(var + kEps);
    ws[OFF_SCALE1 + c] = sc;
    ws[OFF_BIAS1 + c] = beta[c] - mean * sc;
  }
}

// ---------------- K2: fused BN1+ReLU + [g;theta;phi] projection GEMM ----------
// Tile: 128 output channels x 112 positions (2 spatial rows, so the 2x2 pool
// window lives inside the tile). 8 waves; wave w owns M rows [16w,16w+16).
// LDS tiles are fragment-major: A = 8 frags, B = 7 frags.
struct K2Loop {
  alignas(32) __bf16 w[8 * 32 * 16];   // 128x32 A tile, fragment-major
  alignas(32) __bf16 xt[7 * 32 * 16];  // 32x112 B tile, fragment-major
};
union K2Shared { K2Loop loop; float out[128 * 112]; };

__global__ void __launch_bounds__(256) k2_proj(const float* __restrict__ x,
    const float* __restrict__ w_g, const float* __restrict__ b_g,
    const float* __restrict__ w_theta, const float* __restrict__ b_theta,
    const float* __restrict__ w_phi, const float* __restrict__ b_phi,
    float* __restrict__ ws) {
  __shared__ K2Shared sh;
  __shared__ float s1[kC], bi1[kC];
  int b = blockIdx.x / 28, rowpair = blockIdx.x % 28;
  int n0 = rowpair * 112; // rows h=2*rowpair, 2*rowpair+1
  int tid = threadIdx.x, lane = tid & 31, wave = tid >> 5;
  if (tid < kC) { s1[tid] = ws[OFF_SCALE1 + tid]; bi1[tid] = ws[OFF_BIAS1 + tid]; }
  __syncthreads();

  v8f acc[7];
#pragma unroll
  for (int t = 0; t < 7; ++t) acc[t] = vzero8();

  for (int kc = 0; kc < 8; ++kc) {
    int kk = kc * 32;
    // ---- stage A: concat weights (128 x 32) bf16, fragment-major.
    // One thread per (frag,lane) block: 256 blocks == 256 threads.
    {
      int t = tid >> 5, l = tid & 31;
      int m = t * 16 + (l & 15), half = l >> 4;
      const float* wrow;
      if (m < 64)      wrow = w_g + m * kC;
      else if (m < 96) wrow = w_theta + (m - 64) * kC;
      else             wrow = w_phi + (m - 96) * kC;
      v16bf r;
#pragma unroll
      for (int e = 0; e < 16; ++e) r[e] = (__bf16)wrow[kk + a_k_of_e(e, half)];
      *(v16bf*)(sh.loop.w + tid * 16) = r;
    }
    // ---- stage B: x tile with BN1+ReLU (32 K x 112 N), fragment-major.
    // 7*32 = 224 blocks; threads 224..255 prefetch the next K-chunk.
    if (tid < 7 * 32) {
      int f = tid >> 5, l = tid & 31;
      int n = f * 16 + (l & 15), half = l >> 4;
      const float* xp = x + ((size_t)(b * kC + kk + half * 16)) * kHW + n0 + n;
      v16bf r;
#pragma unroll
      for (int e = 0; e < 16; ++e) {
        int ch = kk + half * 16 + e;
        float v = xp[(size_t)e * kHW];
        r[e] = (__bf16)fmaxf(s1[ch] * v + bi1[ch], 0.f);
      }
      *(v16bf*)(sh.loop.xt + tid * 16) = r;
    } else if (kc + 1 < 8) {
      int t = tid - 224; // 32 threads x ~one line each over next chunk
      const float* nx = x + ((size_t)(b * kC + kk + 32 + t)) * kHW + n0;
      __builtin_prefetch(nx, 0, 1);
    }
    __syncthreads();
    v16bf af = ld_frag(sh.loop.w, wave, lane);
#pragma unroll
    for (int nt = 0; nt < 7; ++nt) {
      v16bf bf = ld_frag(sh.loop.xt, nt, lane);
      acc[nt] = wmma_bf16(af, bf, acc[nt]);
    }
    __syncthreads();
  }

  // stage results (+conv bias) into LDS for the fused epilogue
#pragma unroll
  for (int nt = 0; nt < 7; ++nt)
#pragma unroll
    for (int r = 0; r < 8; ++r) {
      int m = wave * 16 + ((lane >> 4) << 3) + r;
      int col = nt * 16 + (lane & 15);
      float bias = (m < 64) ? b_g[m] : ((m < 96) ? b_theta[m - 64] : b_phi[m - 96]);
      sh.out[m * 112 + col] = acc[nt][r] + bias;
    }
  __syncthreads();

  // theta rows 64..95 -> ws (unpooled)
  for (int i = tid; i < kCi * 112; i += 256) {
    int r = i / 112, j = i - r * 112;
    ws[OFF_THETA + ((size_t)(b * kCi + r)) * kHW + n0 + j] = sh.out[(64 + r) * 112 + j];
  }
  // g rows 0..63: fused 2x2 maxpool
  for (int i = tid; i < kCg * 28; i += 256) {
    int r = i / 28, wp = i - r * 28;
    const float* o = &sh.out[r * 112 + 2 * wp];
    float mv = fmaxf(fmaxf(o[0], o[1]), fmaxf(o[56], o[57]));
    ws[OFF_GPOOL + ((size_t)(b * kCg + r)) * kNpPad + rowpair * 28 + wp] = mv;
  }
  // phi rows 96..127: fused 2x2 maxpool
  for (int i = tid; i < kCi * 28; i += 256) {
    int r = i / 28, wp = i - r * 28;
    const float* o = &sh.out[(96 + r) * 112 + 2 * wp];
    float mv = fmaxf(fmaxf(o[0], o[1]), fmaxf(o[56], o[57]));
    ws[OFF_PHIPOOL + ((size_t)(b * kCi + r)) * kNpPad + rowpair * 28 + wp] = mv;
  }
}

// ---------------- K3: M = phi @ g^T / Np, per batch (32x64, K=800 padded) -----
__global__ void __launch_bounds__(256) k3_pg(float* __restrict__ ws) {
  __shared__ alignas(32) __bf16 aP[2 * 32 * 16]; // 32x32 A, fragment-major
  __shared__ alignas(32) __bf16 bG[4 * 32 * 16]; // 32x64 B, fragment-major
  int b = blockIdx.x;
  int tid = threadIdx.x, lane = tid & 31, wave = tid >> 5;
  int mt = wave >> 2, nt = wave & 3;
  v8f acc = vzero8();
  for (int kc = 0; kc < 25; ++kc) {
    int k0 = kc * 32;
    if (tid < 2 * 32) { // A = phi (Ci x Np)
      int t = tid >> 5, l = tid & 31;
      int m = t * 16 + (l & 15), half = l >> 4;
      const float* p = ws + OFF_PHIPOOL + ((size_t)(b * kCi + m)) * kNpPad + k0;
      v16bf r;
#pragma unroll
      for (int e = 0; e < 16; ++e) r[e] = (__bf16)p[a_k_of_e(e, half)];
      *(v16bf*)(aP + tid * 16) = r;
    } else if (tid < 2 * 32 + 4 * 32) { // B[k][cg] = g[cg][k] (transposed read)
      int blk = tid - 64;
      int f = blk >> 5, l = blk & 31;
      int n = f * 16 + (l & 15), half = l >> 4;
      const float* p = ws + OFF_GPOOL + ((size_t)(b * kCg + n)) * kNpPad + k0 + half * 16;
      v16bf r;
#pragma unroll
      for (int e = 0; e < 16; ++e) r[e] = (__bf16)p[e]; // contiguous 16
      *(v16bf*)(bG + blk * 16) = r;
    }
    __syncthreads();
    v16bf af = ld_frag(aP, mt, lane);
    v16bf bf = ld_frag(bG, nt, lane);
    acc = wmma_bf16(af, bf, acc);
    __syncthreads();
  }
  const float inv = 1.0f / (float)kNp;
#pragma unroll
  for (int r = 0; r < 8; ++r) {
    int m = mt * 16 + ((lane >> 4) << 3) + r;
    int n = nt * 16 + (lane & 15);
    ws[OFF_M + ((size_t)(b * kCi + m)) * kCg + n] = acc[r] * inv;
  }
}

// ---------------- K4: y = M^T @ theta (64 x HW per batch) + BN2 partials ------
__global__ void __launch_bounds__(256) k4_y(float* __restrict__ ws) {
  __shared__ alignas(32) __bf16 aM[4 * 32 * 16]; // 64x32 A (M^T), fragment-major
  __shared__ alignas(32) __bf16 tB[4 * 32 * 16]; // 32x64 B (theta), fragment-major
  __shared__ float ylds[64 * 64];
  int b = blockIdx.x / 49, ntile = blockIdx.x % 49;
  int n0 = ntile * 64;
  int tid = threadIdx.x, lane = tid & 31, wave = tid >> 5;
  if (tid < 4 * 32) { // A[cg][ci] = M[ci][cg]
    int t = tid >> 5, l = tid & 31;
    int m = t * 16 + (l & 15), half = l >> 4;
    const float* p = ws + OFF_M + (size_t)b * kCi * kCg + m;
    v16bf r;
#pragma unroll
    for (int e = 0; e < 16; ++e) r[e] = (__bf16)p[(size_t)a_k_of_e(e, half) * kCg];
    *(v16bf*)(aM + tid * 16) = r;
  } else { // B[ci][n] = theta
    int blk = tid - 128;
    int f = blk >> 5, l = blk & 31;
    int n = f * 16 + (l & 15), half = l >> 4;
    const float* p = ws + OFF_THETA + ((size_t)(b * kCi + half * 16)) * kHW + n0 + n;
    v16bf r;
#pragma unroll
    for (int e = 0; e < 16; ++e) r[e] = (__bf16)p[(size_t)e * kHW];
    *(v16bf*)(tB + blk * 16) = r;
  }
  __syncthreads();
  int mt = wave >> 1, ntb = (wave & 1) * 2;
  v16bf af = ld_frag(aM, mt, lane);
#pragma unroll
  for (int t = 0; t < 2; ++t) {
    int nt = ntb + t;
    v16bf bf = ld_frag(tB, nt, lane);
    v8f acc = wmma_bf16(af, bf, vzero8());
#pragma unroll
    for (int r = 0; r < 8; ++r) {
      int m = mt * 16 + ((lane >> 4) << 3) + r;
      int n = nt * 16 + (lane & 15);
      ylds[m * 64 + n] = acc[r];
    }
  }
  __syncthreads();
  for (int i = tid; i < 64 * 64; i += 256) {
    int m = i >> 6, j = i & 63;
    ws[OFF_Y + ((size_t)(b * kCg + m)) * kHW + n0 + j] = ylds[i];
  }
  if (tid < kCg) { // deterministic per-block partial sums (no float atomics)
    float s = 0.f, q = 0.f;
    for (int j = 0; j < 64; ++j) { float v = ylds[tid * 64 + j]; s += v; q += v * v; }
    ws[OFF_PART + (size_t)blockIdx.x * 128 + tid] = s;
    ws[OFF_PART + (size_t)blockIdx.x * 128 + 64 + tid] = q;
  }
}

// ---------------- K5: BN2 finalize --------------------------------------------
__global__ void k5_bn2(const float* __restrict__ g2, const float* __restrict__ bt2,
                       float* __restrict__ ws) {
  int c = threadIdx.x;
  if (c >= kCg) return;
  float s = 0.f, q = 0.f;
  for (int i = 0; i < kB * 49; ++i) {
    s += ws[OFF_PART + (size_t)i * 128 + c];
    q += ws[OFF_PART + (size_t)i * 128 + 64 + c];
  }
  float mean = s / (float)kNtot;
  float var = q / (float)kNtot - mean * mean;
  float sc = g2[c] * rsqrtf(var + kEps);
  ws[OFF_SCALE2 + c] = sc;
  ws[OFF_BIAS2 + c] = bt2[c] - mean * sc;
}

// ---------------- K6: out = w_z @ relu(bn2(y)) + b_z + x ----------------------
__global__ void __launch_bounds__(256) k6_z(const float* __restrict__ x,
    const float* __restrict__ w_z, const float* __restrict__ b_z,
    const float* __restrict__ ws, float* __restrict__ out) {
  // A: w_z 256x64, two K-chunks of 32, each 16 frags, fragment-major
  __shared__ alignas(32) __bf16 wzl[2 * 16 * 32 * 16];
  // B: y tile 64x64 (two K-chunks x 4 N frags), fragment-major
  __shared__ alignas(32) __bf16 yb[2 * 4 * 32 * 16];
  __shared__ float s2[64], bi2[64];
  int b = blockIdx.x / 49, ntile = blockIdx.x % 49;
  int n0 = ntile * 64;
  int tid = threadIdx.x, lane = tid & 31, wave = tid >> 5;
  if (tid < 64) { s2[tid] = ws[OFF_SCALE2 + tid]; bi2[tid] = ws[OFF_BIAS2 + tid]; }
  // stage A: 1024 (kc,frag,lane) blocks, 4 per thread
  for (int blk = tid; blk < 2 * 16 * 32; blk += 256) {
    int kc = blk >> 9, rem = blk & 511;
    int t = rem >> 5, l = rem & 31;
    int m = t * 16 + (l & 15), half = l >> 4;
    const float* p = w_z + m * kCg + kc * 32;
    v16bf r;
#pragma unroll
    for (int e = 0; e < 16; ++e) r[e] = (__bf16)p[a_k_of_e(e, half)];
    *(v16bf*)(wzl + blk * 16) = r;
  }
  __syncthreads(); // s2/bi2 visible
  // stage B: 256 (kc,frag,lane) blocks, 1 per thread; BN2+ReLU applied
  {
    int kc = tid >> 7, rem = tid & 127;
    int f = rem >> 5, l = rem & 31;
    int n = f * 16 + (l & 15), half = l >> 4;
    int ch0 = kc * 32 + half * 16;
    const float* p = ws + OFF_Y + ((size_t)(b * kCg + ch0)) * kHW + n0 + n;
    v16bf r;
#pragma unroll
    for (int e = 0; e < 16; ++e) {
      float v = p[(size_t)e * kHW];
      r[e] = (__bf16)fmaxf(s2[ch0 + e] * v + bi2[ch0 + e], 0.f);
    }
    *(v16bf*)(yb + tid * 16) = r;
  }
  __syncthreads();
  int mt0 = wave * 2;
  v8f acc[2][4];
#pragma unroll
  for (int mi = 0; mi < 2; ++mi)
#pragma unroll
    for (int nt = 0; nt < 4; ++nt) acc[mi][nt] = vzero8();
#pragma unroll
  for (int kc = 0; kc < 2; ++kc) {
    v16bf bf[4];
#pragma unroll
    for (int nt = 0; nt < 4; ++nt) bf[nt] = ld_frag(yb, kc * 4 + nt, lane);
#pragma unroll
    for (int mi = 0; mi < 2; ++mi) {
      v16bf af = ld_frag(wzl, kc * 16 + mt0 + mi, lane);
#pragma unroll
      for (int nt = 0; nt < 4; ++nt) acc[mi][nt] = wmma_bf16(af, bf[nt], acc[mi][nt]);
    }
  }
#pragma unroll
  for (int mi = 0; mi < 2; ++mi)
#pragma unroll
    for (int nt = 0; nt < 4; ++nt)
#pragma unroll
      for (int r = 0; r < 8; ++r) {
        int m = (mt0 + mi) * 16 + ((lane >> 4) << 3) + r;
        int n = n0 + nt * 16 + (lane & 15);
        size_t idx = ((size_t)(b * kC + m)) * kHW + n;
        out[idx] = acc[mi][nt][r] + b_z[m] + x[idx];
      }
}

extern "C" void kernel_launch(void* const* d_in, const int* in_sizes, int n_in,
                              void* d_out, int out_size, void* d_ws, size_t ws_size,
                              hipStream_t stream) {
  (void)in_sizes; (void)n_in; (void)out_size; (void)ws_size;
  const float* x    = (const float*)d_in[0];
  const float* g1   = (const float*)d_in[1];
  const float* be1  = (const float*)d_in[2];
  const float* w_g  = (const float*)d_in[3];
  const float* b_g  = (const float*)d_in[4];
  const float* w_t  = (const float*)d_in[5];
  const float* b_t  = (const float*)d_in[6];
  const float* w_p  = (const float*)d_in[7];
  const float* b_p  = (const float*)d_in[8];
  const float* g2   = (const float*)d_in[9];
  const float* be2  = (const float*)d_in[10];
  const float* w_z  = (const float*)d_in[11];
  const float* b_z  = (const float*)d_in[12];
  float* out = (float*)d_out;
  float* ws  = (float*)d_ws;

  k0_zero<<<(int)((ZERO_LEN + 255) / 256), 256, 0, stream>>>(ws);
  k1_bn1_stats<<<kC, 256, 0, stream>>>(x, g1, be1, ws);
  k2_proj<<<kB * 28, 256, 0, stream>>>(x, w_g, b_g, w_t, b_t, w_p, b_p, ws);
  k3_pg<<<kB, 256, 0, stream>>>(ws);
  k4_y<<<kB * 49, 256, 0, stream>>>(ws);
  k5_bn2<<<1, 64, 0, stream>>>(g2, be2, ws);
  k6_z<<<kB * 49, 256, 0, stream>>>(x, w_z, b_z, ws, out);
}
